// MixtralDecoderLayer_47450798686725
// MI455X (gfx1250) — matrix operand order repaired and checked by reading, not compiled, not run on hardware
//
#include <hip/hip_runtime.h>

// ---------------------------------------------------------------------------
// Types for CDNA5 WMMA (wave32)
// ---------------------------------------------------------------------------
typedef __bf16 v16bf __attribute__((ext_vector_type(16)));
typedef __bf16 bf16x8 __attribute__((ext_vector_type(8)));
typedef float  v8f   __attribute__((ext_vector_type(8)));

#define D_    1024   // hidden dim
#define HD_   64     // head dim
#define NH_   16     // query heads
#define NKV_  8      // kv heads
#define DKV_  512    // NKV_*HD_
#define MI_   3584   // expert intermediate
#define NE_   8      // experts
#define SEQ_  1024
#define TOK_  2048   // B*S

// bf16 <-> f32 (round to nearest even), storage is raw u16
static __device__ __forceinline__ unsigned short f2bf(float f) {
  union { float f; unsigned u; } v; v.f = f;
  unsigned r = (v.u + 0x7fffu + ((v.u >> 16) & 1u)) >> 16;
  return (unsigned short)r;
}
static __device__ __forceinline__ float bf2f(unsigned short h) {
  union { unsigned u; float f; } v; v.u = ((unsigned)h) << 16; return v.f;
}

// ---------------------------------------------------------------------------
// RMSNorm: per-token, fp32 in -> bf16 out (x * rsqrt(mean(x^2)+eps) * w)
// ---------------------------------------------------------------------------
__global__ __launch_bounds__(256) void rmsnorm_kernel(
    const float* __restrict__ x, const float* __restrict__ w,
    unsigned short* __restrict__ out)
{
  int t = blockIdx.x;
  const float* xr = x + (size_t)t * D_;
  __shared__ float red[256];
  int tid = threadIdx.x;
  float s = 0.f;
  for (int i = tid; i < D_; i += 256) { float v = xr[i]; s += v * v; }
  red[tid] = s; __syncthreads();
  for (int o = 128; o > 0; o >>= 1) {
    if (tid < o) red[tid] += red[tid + o];
    __syncthreads();
  }
  float rs = rsqrtf(red[0] * (1.0f / (float)D_) + 1e-6f);
  for (int i = tid; i < D_; i += 256)
    out[(size_t)t * D_ + i] = f2bf(xr[i] * rs * w[i]);
}

// ---------------------------------------------------------------------------
// Generic WMMA GEMM: C[MxN] = A(bf16, row-major, lda) * B(fp32->bf16, row-major, ldb)
// Block tile 128x64, K-step 32, 4 waves. Each wave: 32 rows x 64 cols ->
// 2 A-fragments x 4 B-fragments = 8 wmma/K-step (B frags reused for both A frags).
// Register-stage software pipeline: next k-tile global loads overlap WMMA.
// MODE: 0 = store f32, 1 = store bf16, 2 = store f32 (resid + acc), 3 = f32 +=
// ---------------------------------------------------------------------------
#define BKK 32
#define LDT 40  // LDS row stride in bf16 elems (80B: 16B-aligned runs, bank-spread)

template<int MODE>
__global__ __launch_bounds__(128) void gemm_kernel(
    const unsigned short* __restrict__ A, const float* __restrict__ B,
    float* __restrict__ Cf, unsigned short* __restrict__ Cb,
    const float* __restrict__ resid,
    int M, int N, int K, int lda, int ldb, int ldc)
{
  __shared__ __align__(16) unsigned short As[128 * LDT];
  __shared__ __align__(16) unsigned short Bs[64 * LDT];  // transposed: [n][k]
  const int tid  = threadIdx.x;
  const int lane = tid & 31, w = tid >> 5;
  const int half = lane >> 4, ln = lane & 15;
  const int m0 = blockIdx.y * 128, n0 = blockIdx.x * 64;

  v8f acc[2][4] = {};

  unsigned ar[16];  // A tile: 128x32 bf16 = 2048 u32-pairs
  float    br[16];  // B tile: 32x64 fp32 = 2048 floats

  // prologue: load k-tile 0 into registers
#pragma unroll
  for (int e = 0; e < 16; e++) {
    int idx = tid + 128 * e;
    ar[e] = *(const unsigned*)(A + (size_t)(m0 + (idx >> 4)) * lda + (idx & 15) * 2);
    br[e] = B[(size_t)(idx >> 6) * ldb + n0 + (idx & 63)];
  }

  for (int k0 = 0; k0 < K; k0 += BKK) {
    // registers -> LDS (with fp32->bf16 conversion for B)
#pragma unroll
    for (int e = 0; e < 16; e++) {
      int idx = tid + 128 * e;
      *(unsigned*)(&As[(idx >> 4) * LDT + (idx & 15) * 2]) = ar[e];
      Bs[(idx & 63) * LDT + (idx >> 6)] = f2bf(br[e]);
    }
    __syncthreads();

    // issue next k-tile global loads; results consumed only at next LDS store,
    // so they stay outstanding (LOADcnt) underneath this iteration's WMMAs
    int k1 = k0 + BKK;
    if (k1 < K) {
#pragma unroll
      for (int e = 0; e < 16; e++) {
        int idx = tid + 128 * e;
        ar[e] = *(const unsigned*)(A + (size_t)(m0 + (idx >> 4)) * lda + k1 + (idx & 15) * 2);
        br[e] = B[(size_t)(k1 + (idx >> 6)) * ldb + n0 + (idx & 63)];
      }
    }

    // A fragments: rows w*32 + {0,16} + ln, runs at kbase=(half*8) and kbase+16
    v16bf a[2];
#pragma unroll
    for (int mm = 0; mm < 2; mm++) {
      const unsigned short* ap = &As[(w * 32 + mm * 16 + ln) * LDT + half * 8];
      bf16x8 r0 = *(const bf16x8*)ap;
      bf16x8 r1 = *(const bf16x8*)(ap + 16);
#pragma unroll
      for (int i = 0; i < 8; i++) { a[mm][i] = r0[i]; a[mm][i + 8] = r1[i]; }
    }
#pragma unroll
    for (int t = 0; t < 4; t++) {
      const unsigned short* bp = &Bs[(t * 16 + ln) * LDT + half * 8];
      bf16x8 b0 = *(const bf16x8*)bp;
      bf16x8 b1 = *(const bf16x8*)(bp + 16);
      v16bf b;
#pragma unroll
      for (int i = 0; i < 8; i++) { b[i] = b0[i]; b[i + 8] = b1[i]; }
      acc[0][t] = __builtin_amdgcn_wmma_f32_16x16x32_bf16(
          false, a[0], false, b, (short)0, acc[0][t], false, false);
      acc[1][t] = __builtin_amdgcn_wmma_f32_16x16x32_bf16(
          false, a[1], false, b, (short)0, acc[1][t], false, false);
    }
    __syncthreads();
  }

#pragma unroll
  for (int mm = 0; mm < 2; mm++) {
#pragma unroll
    for (int t = 0; t < 4; t++) {
#pragma unroll
      for (int r = 0; r < 8; r++) {
        int row = m0 + w * 32 + mm * 16 + half * 8 + r;
        int col = n0 + t * 16 + ln;
        size_t o = (size_t)row * ldc + col;
        float v = acc[mm][t][r];
        if (MODE == 0)      Cf[o] = v;
        else if (MODE == 1) Cb[o] = f2bf(v);
        else if (MODE == 2) Cf[o] = resid[o] + v;
        else                Cf[o] += v;
      }
    }
  }
}

// ---------------------------------------------------------------------------
// Fused gate/up GEMM + SwiGLU * routing-weight epilogue -> bf16 act
// Block tile 64x64, 4 waves, 8 wmma/K-step/wave; same register pipeline.
// ---------------------------------------------------------------------------
__global__ __launch_bounds__(128) void gateup_kernel(
    const unsigned short* __restrict__ A, const float* __restrict__ Bg,
    const float* __restrict__ Bu, const float* __restrict__ wts, int e_idx,
    unsigned short* __restrict__ act, int M, int N, int K)
{
  __shared__ __align__(16) unsigned short As[64 * LDT];
  __shared__ __align__(16) unsigned short Bgs[64 * LDT];
  __shared__ __align__(16) unsigned short Bus[64 * LDT];
  const int tid  = threadIdx.x;
  const int lane = tid & 31, w = tid >> 5;
  const int half = lane >> 4, ln = lane & 15;
  const int m0 = blockIdx.y * 64, n0 = blockIdx.x * 64;

  v8f accg[4] = {};
  v8f accu[4] = {};

  unsigned ar[8];   // A tile: 64x32 bf16 = 1024 u32-pairs
  float    brg[16], bru[16];

#pragma unroll
  for (int e = 0; e < 8; e++) {
    int idx = tid + 128 * e;
    ar[e] = *(const unsigned*)(A + (size_t)(m0 + (idx >> 4)) * K + (idx & 15) * 2);
  }
#pragma unroll
  for (int e = 0; e < 16; e++) {
    int idx = tid + 128 * e;
    size_t go = (size_t)(idx >> 6) * N + n0 + (idx & 63);
    brg[e] = Bg[go];
    bru[e] = Bu[go];
  }

  for (int k0 = 0; k0 < K; k0 += BKK) {
#pragma unroll
    for (int e = 0; e < 8; e++) {
      int idx = tid + 128 * e;
      *(unsigned*)(&As[(idx >> 4) * LDT + (idx & 15) * 2]) = ar[e];
    }
#pragma unroll
    for (int e = 0; e < 16; e++) {
      int idx = tid + 128 * e;
      Bgs[(idx & 63) * LDT + (idx >> 6)] = f2bf(brg[e]);
      Bus[(idx & 63) * LDT + (idx >> 6)] = f2bf(bru[e]);
    }
    __syncthreads();

    int k1 = k0 + BKK;
    if (k1 < K) {
#pragma unroll
      for (int e = 0; e < 8; e++) {
        int idx = tid + 128 * e;
        ar[e] = *(const unsigned*)(A + (size_t)(m0 + (idx >> 4)) * K + k1 + (idx & 15) * 2);
      }
#pragma unroll
      for (int e = 0; e < 16; e++) {
        int idx = tid + 128 * e;
        size_t go = (size_t)(k1 + (idx >> 6)) * N + n0 + (idx & 63);
        brg[e] = Bg[go];
        bru[e] = Bu[go];
      }
    }

    const unsigned short* ap = &As[(w * 16 + ln) * LDT + half * 8];
    bf16x8 a0 = *(const bf16x8*)ap;
    bf16x8 a1 = *(const bf16x8*)(ap + 16);
    v16bf a;
#pragma unroll
    for (int i = 0; i < 8; i++) { a[i] = a0[i]; a[i + 8] = a1[i]; }

#pragma unroll
    for (int t = 0; t < 4; t++) {
      const unsigned short* gp = &Bgs[(t * 16 + ln) * LDT + half * 8];
      bf16x8 g0 = *(const bf16x8*)gp;
      bf16x8 g1 = *(const bf16x8*)(gp + 16);
      v16bf bg;
#pragma unroll
      for (int i = 0; i < 8; i++) { bg[i] = g0[i]; bg[i + 8] = g1[i]; }
      accg[t] = __builtin_amdgcn_wmma_f32_16x16x32_bf16(
          false, a, false, bg, (short)0, accg[t], false, false);

      const unsigned short* up = &Bus[(t * 16 + ln) * LDT + half * 8];
      bf16x8 u0 = *(const bf16x8*)up;
      bf16x8 u1 = *(const bf16x8*)(up + 16);
      v16bf bu;
#pragma unroll
      for (int i = 0; i < 8; i++) { bu[i] = u0[i]; bu[i + 8] = u1[i]; }
      accu[t] = __builtin_amdgcn_wmma_f32_16x16x32_bf16(
          false, a, false, bu, (short)0, accu[t], false, false);
    }
    __syncthreads();
  }

#pragma unroll
  for (int t = 0; t < 4; t++) {
#pragma unroll
    for (int r = 0; r < 8; r++) {
      int row = m0 + w * 16 + half * 8 + r;
      int col = n0 + t * 16 + ln;
      float g = accg[t][r], u = accu[t][r];
      float sig = 1.0f / (1.0f + __expf(-g));
      float wgt = wts[(size_t)row * NE_ + e_idx];
      act[(size_t)row * N + col] = f2bf(g * sig * u * wgt);
    }
  }
}

// ---------------------------------------------------------------------------
// RoPE, in-place on bf16 Q and K (one head per 32-thread block)
// ---------------------------------------------------------------------------
__global__ __launch_bounds__(32) void rope_kernel(
    unsigned short* __restrict__ q, unsigned short* __restrict__ k)
{
  int tok  = blockIdx.x;
  int head = blockIdx.y;     // [0,NH_) -> q, [NH_, NH_+NKV_) -> k
  int lane = threadIdx.x;    // 0..31 (= HD_/2)
  int s = tok & (SEQ_ - 1);
  unsigned short* p;
  if (head < NH_) p = q + (size_t)tok * D_   + head * HD_;
  else            p = k + (size_t)tok * DKV_ + (head - NH_) * HD_;
  float inv = __powf(10000.0f, -(float)(2 * lane) / (float)HD_);
  float ang = (float)s * inv;
  float sn, cs;
  __sincosf(ang, &sn, &cs);
  float x1 = bf2f(p[lane]);
  float x2 = bf2f(p[lane + 32]);
  p[lane]      = f2bf(x1 * cs - x2 * sn);
  p[lane + 32] = f2bf(x2 * cs + x1 * sn);
}

// ---------------------------------------------------------------------------
// Flash attention, causal, GQA(rep=2). One wave per (b, h, 16-row q tile).
// QK^T and P*V via v_wmma_f32_16x16x32_bf16; online softmax in f32.
// ---------------------------------------------------------------------------
__global__ __launch_bounds__(32) void attn_kernel(
    const unsigned short* __restrict__ q, const unsigned short* __restrict__ k,
    const unsigned short* __restrict__ v, unsigned short* __restrict__ ao)
{
  __shared__ __align__(16) unsigned short Pl[16 * LDT];  // probs 16x32
  __shared__ __align__(16) unsigned short Vt[64 * LDT];  // V^T: [hd][kpos]
  const int qi = blockIdx.x, h = blockIdx.y, b = blockIdx.z;
  const int kvh  = h >> 1;
  const int lane = threadIdx.x, half = lane >> 4, ln = lane & 15;
  const int kbase = half * 8;

  // Q A-fragments (HD=64 -> two K=32 chunks); contiguous 16B runs from global
  const unsigned short* qrow =
      q + (size_t)(b * SEQ_ + qi * 16 + ln) * D_ + h * HD_;
  v16bf aq0, aq1;
  {
    bf16x8 r0 = *(const bf16x8*)(qrow + kbase);
    bf16x8 r1 = *(const bf16x8*)(qrow + kbase + 16);
#pragma unroll
    for (int i = 0; i < 8; i++) { aq0[i] = r0[i]; aq0[i + 8] = r1[i]; }
    bf16x8 r2 = *(const bf16x8*)(qrow + 32 + kbase);
    bf16x8 r3 = *(const bf16x8*)(qrow + 32 + kbase + 16);
#pragma unroll
    for (int i = 0; i < 8; i++) { aq1[i] = r2[i]; aq1[i + 8] = r3[i]; }
  }

  float m[8], l[8];
  v8f o[4] = {};
#pragma unroll
  for (int r = 0; r < 8; r++) { m[r] = -1e30f; l[r] = 0.f; }

  const int jmax = (qi * 16 + 15) >> 5;
  for (int j = 0; j <= jmax; j++) {
    // scores: two 16x16 C tiles over a 32-wide key block
    v8f s[2];
#pragma unroll
    for (int t = 0; t < 2; t++) {
      const unsigned short* krow =
          k + (size_t)(b * SEQ_ + j * 32 + t * 16 + ln) * DKV_ + kvh * HD_;
      v16bf bk0, bk1;
      {
        bf16x8 r0 = *(const bf16x8*)(krow + half * 8);
        bf16x8 r1 = *(const bf16x8*)(krow + half * 8 + 16);
#pragma unroll
        for (int i = 0; i < 8; i++) { bk0[i] = r0[i]; bk0[i + 8] = r1[i]; }
        bf16x8 r2 = *(const bf16x8*)(krow + 32 + half * 8);
        bf16x8 r3 = *(const bf16x8*)(krow + 32 + half * 8 + 16);
#pragma unroll
        for (int i = 0; i < 8; i++) { bk1[i] = r2[i]; bk1[i + 8] = r3[i]; }
      }
      v8f z = {};
      z = __builtin_amdgcn_wmma_f32_16x16x32_bf16(false, aq0, false, bk0,
                                                  (short)0, z, false, false);
      z = __builtin_amdgcn_wmma_f32_16x16x32_bf16(false, aq1, false, bk1,
                                                  (short)0, z, false, false);
      s[t] = z;
    }

    // online softmax (per row r; 16-lane group reductions)
    float alpha[8];
#pragma unroll
    for (int r = 0; r < 8; r++) {
      int rowg = qi * 16 + half * 8 + r;
      int c0 = j * 32 + ln, c1 = c0 + 16;
      float v0 = s[0][r] * 0.125f;   // 1/sqrt(64)
      float v1 = s[1][r] * 0.125f;
      if (c0 > rowg) v0 = -1e30f;
      if (c1 > rowg) v1 = -1e30f;
      float mx = fmaxf(v0, v1);
#pragma unroll
      for (int d = 8; d >= 1; d >>= 1) mx = fmaxf(mx, __shfl_xor(mx, d, 32));
      float nm = fmaxf(m[r], mx);
      float al = __expf(m[r] - nm);
      float p0 = __expf(v0 - nm), p1 = __expf(v1 - nm);
      float rs = p0 + p1;
#pragma unroll
      for (int d = 8; d >= 1; d >>= 1) rs += __shfl_xor(rs, d, 32);
      m[r] = nm; l[r] = l[r] * al + rs; alpha[r] = al;
      Pl[(half * 8 + r) * LDT + ln]      = f2bf(p0);
      Pl[(half * 8 + r) * LDT + 16 + ln] = f2bf(p1);
    }
#pragma unroll
    for (int t = 0; t < 4; t++)
#pragma unroll
      for (int r = 0; r < 8; r++) o[t][r] *= alpha[r];

    // stage V^T (32 kpos x 64 hd -> Vt[hd][kpos])
    for (int e = 0; e < 64; e++) {
      int idx = lane + 32 * e;
      int row = idx >> 6, hd = idx & 63;
      Vt[hd * LDT + row] =
          v[(size_t)(b * SEQ_ + j * 32 + row) * DKV_ + kvh * HD_ + hd];
    }
    __syncthreads();

    // P as A-fragment, V^T as B-fragments; accumulate into o
    v16bf pf;
    {
      bf16x8 r0 = *(const bf16x8*)(&Pl[ln * LDT + kbase]);
      bf16x8 r1 = *(const bf16x8*)(&Pl[ln * LDT + kbase + 16]);
#pragma unroll
      for (int i = 0; i < 8; i++) { pf[i] = r0[i]; pf[i + 8] = r1[i]; }
    }
#pragma unroll
    for (int t = 0; t < 4; t++) {
      const unsigned short* vp = &Vt[(t * 16 + ln) * LDT + half * 8];
      bf16x8 b0 = *(const bf16x8*)vp;
      bf16x8 b1 = *(const bf16x8*)(vp + 16);
      v16bf bv;
#pragma unroll
      for (int i = 0; i < 8; i++) { bv[i] = b0[i]; bv[i + 8] = b1[i]; }
      o[t] = __builtin_amdgcn_wmma_f32_16x16x32_bf16(false, pf, false, bv,
                                                     (short)0, o[t], false, false);
    }
    __syncthreads();
  }

#pragma unroll
  for (int r = 0; r < 8; r++) {
    float inv = 1.0f / l[r];
    int row = b * SEQ_ + qi * 16 + half * 8 + r;
#pragma unroll
    for (int t = 0; t < 4; t++)
      ao[(size_t)row * D_ + h * HD_ + t * 16 + ln] = f2bf(o[t][r] * inv);
  }
}

// ---------------------------------------------------------------------------
// Router: per-token logits (E=8), softmax, top-2, normalized dense weights
// ---------------------------------------------------------------------------
__global__ __launch_bounds__(32) void router_kernel(
    const unsigned short* __restrict__ h, const float* __restrict__ rw,
    float* __restrict__ wts)
{
  int t = blockIdx.x, lane = threadIdx.x;
  float acc[NE_];
#pragma unroll
  for (int e = 0; e < NE_; e++) acc[e] = 0.f;
  for (int d = lane; d < D_; d += 32) {
    float hv = bf2f(h[(size_t)t * D_ + d]);
    const float* r = rw + (size_t)d * NE_;
#pragma unroll
    for (int e = 0; e < NE_; e++) acc[e] += hv * r[e];
  }
#pragma unroll
  for (int e = 0; e < NE_; e++)
#pragma unroll
    for (int d = 16; d >= 1; d >>= 1) acc[e] += __shfl_xor(acc[e], d, 32);

  if (lane == 0) {
    float mx = acc[0];
#pragma unroll
    for (int e = 1; e < NE_; e++) mx = fmaxf(mx, acc[e]);
    float p[NE_], sum = 0.f;
#pragma unroll
    for (int e = 0; e < NE_; e++) { p[e] = __expf(acc[e] - mx); sum += p[e]; }
#pragma unroll
    for (int e = 0; e < NE_; e++) p[e] /= sum;
    int i0 = 0;
#pragma unroll
    for (int e = 1; e < NE_; e++) if (p[e] > p[i0]) i0 = e;
    int i1 = (i0 == 0) ? 1 : 0;
#pragma unroll
    for (int e = 0; e < NE_; e++) if (e != i0 && p[e] > p[i1]) i1 = e;
    float s2 = p[i0] + p[i1];
#pragma unroll
    for (int e = 0; e < NE_; e++)
      wts[(size_t)t * NE_ + e] =
          (e == i0) ? p[i0] / s2 : ((e == i1) ? p[i1] / s2 : 0.f);
  }
}

// ---------------------------------------------------------------------------
// Host-side launch pipeline
// ---------------------------------------------------------------------------
extern "C" void kernel_launch(void* const* d_in, const int* in_sizes, int n_in,
                              void* d_out, int out_size, void* d_ws, size_t ws_size,
                              hipStream_t stream)
{
  (void)in_sizes; (void)n_in; (void)out_size; (void)ws_size;
  const float* x   = (const float*)d_in[0];
  const float* ln1 = (const float*)d_in[1];
  const float* ln2 = (const float*)d_in[2];
  const float* wq  = (const float*)d_in[3];
  const float* wk  = (const float*)d_in[4];
  const float* wv  = (const float*)d_in[5];
  const float* wo  = (const float*)d_in[6];
  const float* rw  = (const float*)d_in[7];
  const float* wg  = (const float*)d_in[8];
  const float* wu  = (const float*)d_in[9];
  const float* wd  = (const float*)d_in[10];
  float* out = (float*)d_out;

  char* base = (char*)d_ws;
  size_t off = 0;
  auto alloc = [&](size_t bytes) {
    void* p = base + off;
    off = (off + bytes + 255) & ~(size_t)255;
    return p;
  };
  unsigned short* h1  = (unsigned short*)alloc((size_t)TOK_ * D_   * 2);
  unsigned short* qb  = (unsigned short*)alloc((size_t)TOK_ * D_   * 2);
  unsigned short* kb  = (unsigned short*)alloc((size_t)TOK_ * DKV_ * 2);
  unsigned short* vb  = (unsigned short*)alloc((size_t)TOK_ * DKV_ * 2);
  unsigned short* aob = (unsigned short*)alloc((size_t)TOK_ * D_   * 2);
  unsigned short* h2  = (unsigned short*)alloc((size_t)TOK_ * D_   * 2);
  float*          wts = (float*)alloc((size_t)TOK_ * NE_ * 4);
  unsigned short* act = (unsigned short*)alloc((size_t)TOK_ * MI_ * 2);

  // ---- attention block ----
  rmsnorm_kernel<<<TOK_, 256, 0, stream>>>(x, ln1, h1);
  gemm_kernel<1><<<dim3(D_ / 64,   TOK_ / 128), 128, 0, stream>>>(
      h1, wq, nullptr, qb, nullptr, TOK_, D_,   D_, D_, D_,   D_);
  gemm_kernel<1><<<dim3(DKV_ / 64, TOK_ / 128), 128, 0, stream>>>(
      h1, wk, nullptr, kb, nullptr, TOK_, DKV_, D_, D_, DKV_, DKV_);
  gemm_kernel<1><<<dim3(DKV_ / 64, TOK_ / 128), 128, 0, stream>>>(
      h1, wv, nullptr, vb, nullptr, TOK_, DKV_, D_, D_, DKV_, DKV_);
  rope_kernel<<<dim3(TOK_, NH_ + NKV_), 32, 0, stream>>>(qb, kb);
  attn_kernel<<<dim3(SEQ_ / 16, NH_, 2), 32, 0, stream>>>(qb, kb, vb, aob);
  // out = x + ao @ wo
  gemm_kernel<2><<<dim3(D_ / 64, TOK_ / 128), 128, 0, stream>>>(
      aob, wo, out, nullptr, x, TOK_, D_, D_, D_, D_, D_);

  // ---- MoE block ----
  rmsnorm_kernel<<<TOK_, 256, 0, stream>>>(out, ln2, h2);
  router_kernel<<<TOK_, 32, 0, stream>>>(h2, rw, wts);
  for (int e = 0; e < NE_; e++) {
    gateup_kernel<<<dim3(MI_ / 64, TOK_ / 64), 128, 0, stream>>>(
        h2, wg + (size_t)e * D_ * MI_, wu + (size_t)e * D_ * MI_,
        wts, e, act, TOK_, MI_, D_);
    // out += act @ w_down[e]
    gemm_kernel<3><<<dim3(D_ / 64, TOK_ / 128), 128, 0, stream>>>(
        act, wd + (size_t)e * MI_ * D_, out, nullptr, nullptr,
        TOK_, D_, MI_, MI_, D_, D_);
  }
}
